// GeometryAwareAssign_8074538517112
// MI455X (gfx1250) — matrix operand order repaired and checked by reading, compiled, not run on hardware
//
#include <hip/hip_runtime.h>
#include <math.h>

// ---------------------------------------------------------------------------
// GeometryAwareAssign (SimOTA lane assignment) for MI455X / gfx1250 (wave32).
//
// iou(n,m) = (30*cnt - L1) / (30*cnt + L1 + 1e-9)
//   cnt = sum_s validP(n,s)*validG(m,s)  --> matmul over S: v_wmma_f32_16x16x32_f16
//         (0/1 f16 operands built branchlessly from per-row validity bitmasks)
//   L1  = sum_s valid * |p_xs - g_xs|    --> VALU, float4 LDS, WMMA C/D layout
// ---------------------------------------------------------------------------

typedef __attribute__((ext_vector_type(16))) _Float16 v16h;
typedef __attribute__((ext_vector_type(8)))  float    v8f;

constexpr int   B_      = 16;
constexpr int   N_      = 1000;
constexpr int   M_      = 24;
constexpr int   MPAD    = 32;      // GT rows padded to 32 (tile-1 is branchless)
constexpr int   S_      = 72;
constexpr int   C_      = 78;
constexpr int   KSEL    = 10;
constexpr int   NTILE   = 128;     // preds per block (8 waves x 16 preds)
constexpr int   XSTRIDE = 76;      // LDS row stride: 16B-aligned rows, 76%64=12
constexpr float INVALID = -100000.0f;
constexpr float RAD     = 0.017453292519943295f;
constexpr float BIGF    = 3.4e38f;

__device__ __forceinline__ float cls_cost_from_logits(float l0, float l1) {
    // -log(clip(softmax[1], 1e-8)) = min(softplus(l0-l1), -log(1e-8))
    float d  = l0 - l1;
    float sp = fmaxf(d, 0.0f) + log1pf(expf(-fabsf(d)));
    return fminf(sp, 18.420680743952367f);
}

__device__ __forceinline__ float clamped_tan_deg(float th) {
    float t = tanf(th * RAD);
    return fminf(fmaxf(t, -1000.0f), 1000.0f);
}

// ============================ Kernel 1: pair costs ==========================
__global__ __launch_bounds__(256)
void pair_cost_kernel(const float* __restrict__ preds,
                      const float* __restrict__ targets,
                      const int*   __restrict__ masks,
                      const int*   __restrict__ p_imgw,
                      const int*   __restrict__ p_imgh,
                      float*       __restrict__ iou_out,
                      float*       __restrict__ tot_out,
                      unsigned int* __restrict__ match_bits)
{
    __shared__ __align__(16) float        s_gxs[MPAD][XSTRIDE];
    __shared__ __align__(16) float        s_pxs[NTILE][XSTRIDE];
    __shared__ __align__(16) unsigned int s_pvalid[NTILE][4];   // 72-bit masks
    __shared__ __align__(16) unsigned int s_gvalid[MPAD][4];
    __shared__ float s_gsx[M_], s_gsy[M_], s_gth[M_];
    __shared__ int   s_gmask[M_];
    __shared__ float s_psx[NTILE], s_psy[NTILE], s_pth[NTILE], s_pcls[NTILE];

    const int   b      = blockIdx.y;
    const int   n0     = blockIdx.x * NTILE;
    const int   tid    = threadIdx.x;
    const float img_w  = (float)(*p_imgw);
    const float img_h  = (float)(*p_imgh);
    const float inv_s1 = 1.0f / (float)(S_ - 1);

    // ---------------- fill phase: one thread owns one row -------------------
    if (tid < NTILE) {                       // pred row `tid`
        const int n = n0 + tid;
        unsigned int mk[3] = {0u, 0u, 0u};
        if (n < N_) {
            const float* p = preds + ((size_t)b * N_ + n) * C_;
            const float sy = p[2], sx = p[3], th = p[4], len = p[5];
            const float tn = clamped_tan_deg(th);
            s_psy[tid] = sy; s_psx[tid] = sx; s_pth[tid] = th;
            s_pcls[tid] = cls_cost_from_logits(p[0], p[1]);
            match_bits[b * N_ + n] = 0u;
            for (int s0 = 0; s0 < S_; s0 += 4) {
                float x4[4];
                #pragma unroll
                for (int j = 0; j < 4; ++j) {
                    const int s = s0 + j;
                    const float ys = (img_h - 1.0f) * (float)(S_ - 1 - s) * inv_s1;
                    float x = INVALID;
                    if (ys >= sy - len && ys <= sy) {
                        x = sx + (sy - ys) * tn + p[6 + s];
                        mk[s >> 5] |= 1u << (s & 31);
                    }
                    x4[j] = x;
                }
                *(float4*)&s_pxs[tid][s0] = make_float4(x4[0], x4[1], x4[2], x4[3]);
            }
        } else {
            s_psy[tid] = 0.f; s_psx[tid] = 0.f; s_pth[tid] = 0.f; s_pcls[tid] = 0.f;
            for (int s0 = 0; s0 < S_; s0 += 4)
                *(float4*)&s_pxs[tid][s0] = make_float4(INVALID, INVALID, INVALID, INVALID);
        }
        s_pvalid[tid][0] = mk[0]; s_pvalid[tid][1] = mk[1];
        s_pvalid[tid][2] = mk[2]; s_pvalid[tid][3] = 0u;
    } else if (tid < NTILE + MPAD) {         // GT row `tid - NTILE` (incl. pad)
        const int m = tid - NTILE;
        unsigned int mk[3] = {0u, 0u, 0u};
        if (m < M_) {
            const float* t = targets + ((size_t)b * M_ + m) * C_;
            const float gy = t[2], gx = t[3], gth = t[4];
            const float tn = clamped_tan_deg(gth);
            s_gsy[m] = gy; s_gsx[m] = gx; s_gth[m] = gth;
            s_gmask[m] = masks[b * M_ + m];
            for (int s0 = 0; s0 < S_; s0 += 4) {
                float x4[4];
                #pragma unroll
                for (int j = 0; j < 4; ++j) {
                    const int s = s0 + j;
                    const float ys = (img_h - 1.0f) * (float)(S_ - 1 - s) * inv_s1;
                    const float dx = t[6 + s];
                    float x = gx + (gy - ys) * tn + dx;
                    if (dx < -10000.0f || x < 0.0f || x >= img_w) x = INVALID;
                    else mk[s >> 5] |= 1u << (s & 31);
                    x4[j] = x;
                }
                *(float4*)&s_gxs[m][s0] = make_float4(x4[0], x4[1], x4[2], x4[3]);
            }
        } else {                              // pad rows 24..31: always invalid
            for (int s0 = 0; s0 < S_; s0 += 4)
                *(float4*)&s_gxs[m][s0] = make_float4(INVALID, INVALID, INVALID, INVALID);
        }
        s_gvalid[m][0] = mk[0]; s_gvalid[m][1] = mk[1];
        s_gvalid[m][2] = mk[2]; s_gvalid[m][3] = 0u;
    }
    __syncthreads();

    const int wave = tid >> 5;
    const int lane = tid & 31;
    const int hi   = lane >> 4;         // 0|1
    const int lo   = lane & 15;
    const int g0   = lo;                // gt column, tile 0
    const int g1   = 16 + lo;           // gt column, tile 1 (rows 24..31 padded)
    const int arow = wave * 16 + lo;    // A-matrix row = pred index in tile

    // ---- valid-count matmul: D = validP(16xS) * validG(Sx16) via WMMA ------
    // Operands are 0/1 f16 built from bitmasks; bits >= S are zero by
    // construction, so no bounds checks and EXEC stays uniform throughout.
    const unsigned int pv[3]  = { s_pvalid[arow][0], s_pvalid[arow][1], s_pvalid[arow][2] };
    const unsigned int gva[3] = { s_gvalid[g0][0],   s_gvalid[g0][1],   s_gvalid[g0][2]   };
    const unsigned int gvb[3] = { s_gvalid[g1][0],   s_gvalid[g1][1],   s_gvalid[g1][2]   };

    v8f c0 = {}; v8f c1 = {};
    #pragma unroll
    for (int kc = 0; kc < 3; ++kc) {
        const unsigned int pA = pv[kc]  >> (hi * 8);    // A: K = hi*8 + (e>>3)*16 + (e&7)
        const unsigned int gA = gva[kc] >> (hi * 8);    // B: K = hi*8 + (e>>3)*16 + (e&7)
        const unsigned int gB = gvb[kc] >> (hi * 8);
        v16h a, b0v, b1v;
        #pragma unroll
        for (int e = 0; e < 16; ++e) {
            const int sh = ((e >> 3) << 4) + (e & 7);
            a[e]   = (_Float16)((pA >> sh) & 1u);
            b0v[e] = (_Float16)((gA >> sh) & 1u);
            b1v[e] = (_Float16)((gB >> sh) & 1u);
        }
        c0 = __builtin_amdgcn_wmma_f32_16x16x32_f16(false, a, false, b0v, (short)0, c0, false, false);
        c1 = __builtin_amdgcn_wmma_f32_16x16x32_f16(false, a, false, b1v, (short)0, c1, false, false);
    }

    // ---- L1 term in the WMMA C/D layout: row = v + 8*hi, col = lo ----------
    // s-chunk outer so the two GT rows are loaded once per chunk (float4 DS).
    float l10[8], l11[8];
    #pragma unroll
    for (int v = 0; v < 8; ++v) { l10[v] = 0.f; l11[v] = 0.f; }

    for (int s0 = 0; s0 < S_; s0 += 4) {
        const float4 a4 = *(const float4*)&s_gxs[g0][s0];
        const float4 b4 = *(const float4*)&s_gxs[g1][s0];
        const float ga[4] = { a4.x, a4.y, a4.z, a4.w };
        const float gb[4] = { b4.x, b4.y, b4.z, b4.w };
        #pragma unroll
        for (int v = 0; v < 8; ++v) {
            const int pr = wave * 16 + v + hi * 8;
            const float4 p4 = *(const float4*)&s_pxs[pr][s0];
            const float pp[4] = { p4.x, p4.y, p4.z, p4.w };
            #pragma unroll
            for (int j = 0; j < 4; ++j) {
                const bool pvld = pp[j] != INVALID;
                l10[v] += (pvld && ga[j] != INVALID) ? fabsf(pp[j] - ga[j]) : 0.0f;
                l11[v] += (pvld && gb[j] != INVALID) ? fabsf(pp[j] - gb[j]) : 0.0f;
            }
        }
    }

    // ---- combine & emit iou / total cost -----------------------------------
    #pragma unroll
    for (int v = 0; v < 8; ++v) {
        const int pr = wave * 16 + v + hi * 8;
        const int n  = n0 + pr;
        if (n >= N_) continue;
        const float pcls = s_pcls[pr], psx = s_psx[pr], psy = s_psy[pr], pth = s_pth[pr];
        #pragma unroll
        for (int t = 0; t < 2; ++t) {
            const int m = t ? g1 : g0;
            if (m >= M_) continue;
            const float cnt = t ? c1[v] : c0[v];
            const float l1  = t ? l11[v] : l10[v];
            const float O   = 30.0f * cnt - l1;
            const float U   = 30.0f * cnt + l1;
            float iou = O / (U + 1e-9f);
            const int gm = s_gmask[m];
            iou = gm ? iou : 0.0f;                        // ious masked
            const float dxn  = (psx - s_gsx[m]) / (img_w - 1.0f);
            const float dyn2 = (psy - s_gsy[m]) / (img_h - 1.0f);
            const float dist = sqrtf(dxn * dxn + dyn2 * dyn2 + 1e-8f);
            const float thc  = fabsf((pth - s_gth[m]) / 90.0f);
            float tot = 4.0f * pcls + 5.0f * (dist + 2.0f * thc) + 2.0f * (1.0f - iou);
            if (!gm) tot = 100000.0f;
            const size_t off = ((size_t)b * N_ + n) * M_ + m;
            iou_out[off] = iou;
            tot_out[off] = tot;
        }
    }
}

// ===================== Kernel 2: per-GT dynamic top-k =======================
__device__ __forceinline__ void argmax_reduce(float* rv, int* ri, int tid, float v, int i) {
    rv[tid] = v; ri[tid] = i;
    __syncthreads();
    for (int off = 128; off > 0; off >>= 1) {
        if (tid < off) {
            float ov = rv[tid + off]; int oi = ri[tid + off];
            if (ov > rv[tid] || (ov == rv[tid] && oi < ri[tid])) { rv[tid] = ov; ri[tid] = oi; }
        }
        __syncthreads();
    }
}

__global__ __launch_bounds__(256)
void select_kernel(const float* __restrict__ iou_in,
                   const float* __restrict__ tot_in,
                   const int*   __restrict__ masks,
                   unsigned int* __restrict__ match_bits)
{
    const int bm = blockIdx.x;
    const int b  = bm / M_, m = bm % M_;
    if (!masks[b * M_ + m]) return;      // uniform branch: whole block exits

    __shared__ float vals[N_];
    __shared__ float rv[256];
    __shared__ int   ri[256];
    __shared__ int   s_count;
    const int tid = threadIdx.x;

    // ---- dynamic_k = clip(int(sum top-10 iou), 1, N) ----
    for (int n = tid; n < N_; n += 256)
        vals[n] = iou_in[((size_t)b * N_ + n) * M_ + m];
    __syncthreads();

    float sum0 = 0.0f;                    // meaningful on tid 0 only
    for (int k = 0; k < KSEL; ++k) {
        float bv = -BIGF; int bi = N_;
        for (int n = tid; n < N_; n += 256) {
            float v = vals[n];
            if (v > bv) { bv = v; bi = n; }
        }
        argmax_reduce(rv, ri, tid, bv, bi);
        if (tid == 0) { sum0 += rv[0]; vals[ri[0]] = -BIGF; }
        __syncthreads();
    }
    if (tid == 0) {
        int dk = (int)sum0;               // trunc toward zero, like astype(int32)
        dk = dk < 1 ? 1 : (dk > N_ ? N_ : dk);
        s_count = dk < KSEL ? dk : KSEL;
    }
    __syncthreads();

    // ---- pick the s_count lowest-cost preds for this GT ----
    for (int n = tid; n < N_; n += 256)
        vals[n] = tot_in[((size_t)b * N_ + n) * M_ + m];
    __syncthreads();

    for (int k = 0; k < KSEL; ++k) {
        float bv = BIGF; int bi = N_;
        for (int n = tid; n < N_; n += 256) {
            float v = vals[n];
            if (v < bv) { bv = v; bi = n; }
        }
        argmax_reduce(rv, ri, tid, -bv, bi);   // argmin via negation, idx tie-break
        if (tid == 0) {
            int idx = ri[0];
            vals[idx] = BIGF;
            if (k < s_count)
                atomicOr(&match_bits[b * N_ + idx], 1u << m);
        }
        __syncthreads();
    }
}

// ===================== Kernel 3: conflict resolution ========================
__global__ __launch_bounds__(256)
void resolve_kernel(const float* __restrict__ tot_in,
                    const unsigned int* __restrict__ match_bits,
                    float* __restrict__ out)
{
    const int bn = blockIdx.x * blockDim.x + threadIdx.x;
    if (bn >= B_ * N_) return;
    unsigned int bits = match_bits[bn];
    int pop = __popc(bits);
    int matched = -1;
    if (pop == 1) {
        matched = __ffs(bits) - 1;
    } else if (pop > 1) {
        const float* t = tot_in + (size_t)bn * M_;
        float best = BIGF;
        for (int m = 0; m < M_; ++m) {
            if ((bits >> m) & 1u) {
                float c = t[m];
                if (c < best) { best = c; matched = m; }   // first-min tie-break
            }
        }
    }
    out[bn]            = pop > 0 ? 1.0f : 0.0f;   // assigned
    out[B_ * N_ + bn]  = (float)matched;          // matched gt idx or -1
}

// =============================== launcher ===================================
extern "C" void kernel_launch(void* const* d_in, const int* in_sizes, int n_in,
                              void* d_out, int out_size, void* d_ws, size_t ws_size,
                              hipStream_t stream)
{
    (void)in_sizes; (void)n_in; (void)out_size; (void)ws_size;
    const float* preds   = (const float*)d_in[0];
    const float* targets = (const float*)d_in[1];
    const int*   masks   = (const int*)d_in[2];
    const int*   imgw    = (const int*)d_in[3];
    const int*   imgh    = (const int*)d_in[4];

    float* ws   = (float*)d_ws;
    float* iou  = ws;                                   // B*N*M f32
    float* tot  = ws + (size_t)B_ * N_ * M_;            // B*N*M f32
    unsigned int* bits = (unsigned int*)(ws + (size_t)2 * B_ * N_ * M_);  // B*N u32
    float* out  = (float*)d_out;

    dim3 g1((N_ + NTILE - 1) / NTILE, B_);
    pair_cost_kernel<<<g1, 256, 0, stream>>>(preds, targets, masks, imgw, imgh, iou, tot, bits);
    select_kernel<<<B_ * M_, 256, 0, stream>>>(iou, tot, masks, bits);
    resolve_kernel<<<(B_ * N_ + 255) / 256, 256, 0, stream>>>(tot, bits, out);
}
